// DilatedMDTA_75316546503094
// MI455X (gfx1250) — matrix operand order repaired and checked by reading, not compiled, not run on hardware
//
#include <hip/hip_runtime.h>

#define HW 16384

typedef __attribute__((ext_vector_type(16))) __bf16        v16bf;
typedef __attribute__((ext_vector_type(2)))  __bf16        v2bf;
typedef __attribute__((ext_vector_type(8)))  float         v8f;
typedef __attribute__((ext_vector_type(4)))  float         f32x4;
typedef __attribute__((ext_vector_type(2)))  float         f32x2;
typedef __attribute__((ext_vector_type(4)))  unsigned int  u32x4;
typedef __attribute__((ext_vector_type(8)))  int           i32x8;
typedef __attribute__((ext_vector_type(4)))  int           i32x4;

union BFrag {
    unsigned short u[16];
    v2bf           h[8];
    u32x4          q[2];
    v16bf          v;
};

#if __has_builtin(__builtin_amdgcn_cvt_pk_bf16_f32)
#define HAVE_PK_BF16 1
#endif
#if defined(__gfx1250__) && __has_builtin(__builtin_amdgcn_tensor_load_to_lds)
#define USE_TDM 1
#endif

// convert 4 floats into BFrag bf16 pair-slots h[pos], h[pos+1] (u[2p..2p+3])
__device__ __forceinline__ void cvt4(BFrag& d, int pos, f32x4 v) {
#ifdef HAVE_PK_BF16
    d.h[pos]     = __builtin_amdgcn_cvt_pk_bf16_f32(v[0], v[1]);
    d.h[pos + 1] = __builtin_amdgcn_cvt_pk_bf16_f32(v[2], v[3]);
#else
    // fptrunc float->bfloat: lowers to v_cvt_pk_bf16_f32 when the target has
    // it, otherwise clang's RNE emulation (same cost as manual bit tricks).
    v2bf p0, p1;
    p0[0] = (__bf16)v[0];
    p0[1] = (__bf16)v[1];
    p1[0] = (__bf16)v[2];
    p1[1] = (__bf16)v[3];
    d.h[pos]     = p0;
    d.h[pos + 1] = p1;
#endif
}

#ifdef USE_TDM
// Tensor Data Mover: DMA a 32-row x 128-col f32 tile of a (192 x 16384) tensor
// (row stride 16384 elems) from global into LDS, row-major. ISA ch.8 D# layout.
__device__ __forceinline__ void tdm_load_tile(const float* gsrc, const float* ldst) {
    const unsigned long long ga = (unsigned long long)(uintptr_t)gsrc;
    const unsigned la = (unsigned)(uintptr_t)ldst;   // LDS aperture: low 32 bits
    u32x4 g0;
    g0[0] = 1u;                                      // count=1, no gather/iterate
    g0[1] = la;                                      // lds_addr
    g0[2] = (unsigned)ga;                            // global_addr[31:0]
    g0[3] = (unsigned)((ga >> 32) & 0x01FFFFFFu) | (2u << 30);  // addr[56:32]|type=2
    i32x8 g1;
    g1[0] = 0x00020000;                              // data_size = 4 bytes
    g1[1] = (int)(16384u << 16);                     // tensor_dim0 = 16384
    g1[2] = (int)(192u << 16);                       // tensor_dim1 = 192
    g1[3] = (int)(128u << 16);                       // tile_dim0 = 128
    g1[4] = 32;                                      // tile_dim1 = 32
    g1[5] = 16384;                                   // tensor_dim0_stride
    g1[6] = 0;
    g1[7] = 0;
    const i32x4 z4 = {0, 0, 0, 0};
    const i32x8 z8 = {0, 0, 0, 0, 0, 0, 0, 0};
    __builtin_amdgcn_tensor_load_to_lds(g0, g1, z4, z4, z8, 0);
}
#endif

// ---------------------------------------------------------------------------
// C[b] (O x 16384) = W (O x K) * X[b] (K x 16384), K == 192 (template, fully
// unrolled). Block tile 64x128, 8 waves, wave = 16x64 strip (4 accumulators).
// X K-chunks staged to LDS: TDM DMA (double buffered) -> cvt to bf16,
// stored transposed so each lane's B fragment is two contiguous ds_load_b128.
// ---------------------------------------------------------------------------
template <int K>
__global__ __launch_bounds__(256)
void gemm_wmma_bf16(const float* __restrict__ W, const float* __restrict__ X,
                    float* __restrict__ C) {
    constexpr int NS = K / 32;
    const int pBlock = blockIdx.x * 128;
    const int mBlock = blockIdx.y * 64;
    const float* Xb = X + (size_t)blockIdx.z * (size_t)K * HW;
    float*       Cb = C + (size_t)blockIdx.z * (size_t)gridDim.y * 64 * HW;

    __shared__ __align__(16) unsigned short Xs[128][40];  // 32 K + 8 pad (80B row)
#ifdef USE_TDM
    __shared__ __align__(16) float fbuf[2][32][128];      // TDM double buffer
#endif

    const int t    = threadIdx.x;
    const int lane = t & 31;
    const int wid  = t >> 5;
    const int wrow = wid & 3;
    const int wcol = wid >> 2;
    const int r    = lane & 15;
    const int hi   = lane >> 4;

    const int kRow = t >> 3;          // 0..31
    const int pSub = (t & 7) * 16;    // 0..112

#ifdef USE_TDM
    const bool leader = (__builtin_amdgcn_readfirstlane(t) == 0);  // wave 0 only
    if (leader) tdm_load_tile(Xb + pBlock, &fbuf[0][0][0]);
#endif

    v8f acc[4] = {};

#pragma unroll
    for (int s = 0; s < NS; s++) {
        const int kc = s * 32;
#ifdef USE_TDM
        if (leader) __builtin_amdgcn_s_wait_tensorcnt(0);  // fbuf[s&1] ready
        __syncthreads();   // publish tile; previous frag reads of Xs complete
        if (leader && (s + 1 < NS))
            tdm_load_tile(Xb + (size_t)(kc + 32) * HW + pBlock,
                          &fbuf[(s + 1) & 1][0][0]);
        const float* src = &fbuf[s & 1][kRow][pSub];
#else
        __syncthreads();   // previous frag reads of Xs complete
        const float* src = Xb + (size_t)(kc + kRow) * HW + pBlock + pSub;
        if (s + 1 < NS) __builtin_prefetch(src + (size_t)32 * HW, 0, 1);
#endif
        f32x4 x0 = ((const f32x4*)src)[0];
        f32x4 x1 = ((const f32x4*)src)[1];
        f32x4 x2 = ((const f32x4*)src)[2];
        f32x4 x3 = ((const f32x4*)src)[3];
        BFrag tf;
        cvt4(tf, 0, x0); cvt4(tf, 2, x1); cvt4(tf, 4, x2); cvt4(tf, 6, x3);
#pragma unroll
        for (int i = 0; i < 16; i++) Xs[pSub + i][kRow] = tf.u[i];
        __syncthreads();

        // A fragment: row M = mBlock + wrow*16 + r, K = hi*16 + {0..15}
        const float* wp = W + (size_t)(mBlock + wrow * 16 + r) * K + kc + hi * 16;
        f32x4 a0 = ((const f32x4*)wp)[0];
        f32x4 a1 = ((const f32x4*)wp)[1];
        f32x4 a2 = ((const f32x4*)wp)[2];
        f32x4 a3 = ((const f32x4*)wp)[3];
        BFrag af;
        cvt4(af, 0, a0); cvt4(af, 2, a1); cvt4(af, 4, a2); cvt4(af, 6, a3);

        // B fragments from LDS (contiguous per lane thanks to transpose)
#pragma unroll
        for (int nt = 0; nt < 4; nt++) {
            const unsigned short* bp = &Xs[wcol * 64 + nt * 16 + r][hi * 16];
            BFrag bf;
            bf.q[0] = ((const u32x4*)bp)[0];
            bf.q[1] = ((const u32x4*)bp)[1];
            acc[nt] = __builtin_amdgcn_wmma_f32_16x16x32_bf16(
                false, af.v, false, bf.v, (short)0, acc[nt], false, false);
        }
    }

    // store (C layout: VGPR r8 -> row r8 + hi*8, col = lane%16)
#pragma unroll
    for (int nt = 0; nt < 4; nt++) {
        const int col = pBlock + wcol * 64 + nt * 16 + r;
#pragma unroll
        for (int r8 = 0; r8 < 8; r8++) {
            const int row = mBlock + wrow * 16 + r8 + hi * 8;
            Cb[(size_t)row * HW + col] = acc[nt][r8];
        }
    }
}

// ---------------------------------------------------------------------------
// Depthwise 3x3, dilation 2, pad 2. One thread per output pixel.
// ---------------------------------------------------------------------------
__global__ __launch_bounds__(256)
void dwconv_kernel(const float* __restrict__ in, const float* __restrict__ wdw,
                   float* __restrict__ out) {
    const int bo  = blockIdx.y;                      // b*576 + o
    const int o   = bo % 576;
    const int pix = blockIdx.x * 256 + threadIdx.x;  // 0..16383
    const int y = pix >> 7, x = pix & 127;
    const float* ip = in + (size_t)bo * HW;
    const float* wp = wdw + o * 9;
    float s = 0.f;
#pragma unroll
    for (int i = -1; i <= 1; i++) {
        const int yy = y + 2 * i;
        if (yy < 0 || yy > 127) continue;
#pragma unroll
        for (int j = -1; j <= 1; j++) {
            const int xx = x + 2 * j;
            if (xx < 0 || xx > 127) continue;
            s += ip[yy * 128 + xx] * wp[(i + 1) * 3 + (j + 1)];
        }
    }
    out[(size_t)bo * HW + pix] = s;
}

// ---------------------------------------------------------------------------
// scale = 1 / (||row||_2 + eps) for q planes (block 0..1535) and k planes.
// ---------------------------------------------------------------------------
__global__ __launch_bounds__(256)
void norm_kernel(const float* __restrict__ qkvdw, float* __restrict__ scales) {
    const int p   = blockIdx.x;
    const int isK = (p >= 1536) ? 1 : 0;
    const int pc  = p - isK * 1536;
    const int b = pc / 192, c = pc % 192;
    const float* row = qkvdw + ((size_t)b * 576 + isK * 192 + c) * HW;
    float s = 0.f;
    for (int i = threadIdx.x * 4; i < HW; i += 1024) {
        f32x4 v = *(const f32x4*)(row + i);
        s += v[0] * v[0] + v[1] * v[1] + v[2] * v[2] + v[3] * v[3];
    }
    __shared__ float red[256];
    red[threadIdx.x] = s;
    __syncthreads();
    for (int st = 128; st > 0; st >>= 1) {
        if (threadIdx.x < st) red[threadIdx.x] += red[threadIdx.x + st];
        __syncthreads();
    }
    if (threadIdx.x == 0) scales[p] = 1.f / (sqrtf(red[0]) + 1e-6f);
}

// ---------------------------------------------------------------------------
// attn[b,h] = softmax(scale_q * (q k^T) * scale_k * temperature).
// One block per (b,head); 8 waves split K=16384; 2x2 WMMA tiles (24 -> 32 pad).
// ---------------------------------------------------------------------------
__global__ __launch_bounds__(256)
void attn_softmax_kernel(const float* __restrict__ qkvdw,
                         const float* __restrict__ scq,
                         const float* __restrict__ sck,
                         const float* __restrict__ temp,
                         float* __restrict__ attn) {
    const int bh = blockIdx.x;
    const int b = bh >> 3, head = bh & 7;
    const float* qp = qkvdw + ((size_t)b * 576 + head * 24) * HW;
    const float* kp = qkvdw + ((size_t)b * 576 + 192 + head * 24) * HW;

    const int t = threadIdx.x, lane = t & 31, wid = t >> 5;
    const int r = lane & 15, hi = lane >> 4;

    v8f acc[2][2] = {};

    for (int n0 = wid * 32; n0 < HW; n0 += 256) {
        BFrag af[2], bf[2];
#pragma unroll
        for (int mt = 0; mt < 2; mt++) {
            const int c = mt * 16 + r;
#pragma unroll
            for (int i = 0; i < 16; i++) af[mt].u[i] = 0;
            if (c < 24) {
                const float* p = qp + (size_t)c * HW + n0 + hi * 16;
                cvt4(af[mt], 0, ((const f32x4*)p)[0]);
                cvt4(af[mt], 2, ((const f32x4*)p)[1]);
                cvt4(af[mt], 4, ((const f32x4*)p)[2]);
                cvt4(af[mt], 6, ((const f32x4*)p)[3]);
            }
        }
#pragma unroll
        for (int nt = 0; nt < 2; nt++) {
            const int d = nt * 16 + r;
#pragma unroll
            for (int i = 0; i < 16; i++) bf[nt].u[i] = 0;
            if (d < 24) {
                const float* p = kp + (size_t)d * HW + n0 + hi * 16;
#pragma unroll
                for (int j = 0; j < 4; j++)
                    cvt4(bf[nt], 2 * j, ((const f32x4*)p)[j]);
            }
        }
#pragma unroll
        for (int mt = 0; mt < 2; mt++)
#pragma unroll
            for (int nt = 0; nt < 2; nt++)
                acc[mt][nt] = __builtin_amdgcn_wmma_f32_16x16x32_bf16(
                    false, af[mt].v, false, bf[nt].v, (short)0, acc[mt][nt],
                    false, false);
    }

    __shared__ float red[1024];
    for (int i = t; i < 1024; i += 256) red[i] = 0.f;
    __syncthreads();
#pragma unroll
    for (int mt = 0; mt < 2; mt++)
#pragma unroll
        for (int nt = 0; nt < 2; nt++)
#pragma unroll
            for (int r8 = 0; r8 < 8; r8++)
                atomicAdd(&red[(mt * 16 + r8 + hi * 8) * 32 + nt * 16 + r],
                          acc[mt][nt][r8]);
    __syncthreads();

    if (t < 32) {
        float* outp = attn + (size_t)bh * 1024 + t * 32;
        if (t < 24) {
            const float sq = scq[b * 192 + head * 24 + t] * temp[head];
            float vrow[24];
            float mx = -1e30f;
#pragma unroll
            for (int d = 0; d < 24; d++) {
                vrow[d] = red[t * 32 + d] * sq * sck[b * 192 + head * 24 + d];
                mx = fmaxf(mx, vrow[d]);
            }
            float sum = 0.f;
#pragma unroll
            for (int d = 0; d < 24; d++) {
                vrow[d] = __expf(vrow[d] - mx);
                sum += vrow[d];
            }
            const float inv = 1.f / sum;
#pragma unroll
            for (int d = 0; d < 24; d++) outp[d] = vrow[d] * inv;
#pragma unroll
            for (int d = 24; d < 32; d++) outp[d] = 0.f;
        } else {
#pragma unroll
            for (int d = 0; d < 32; d++) outp[d] = 0.f;
        }
    }
}

// ---------------------------------------------------------------------------
// out[b,h] (24 x 16384) = attn (24x24) * v (24 x 16384). Streaming VALU.
// ---------------------------------------------------------------------------
__global__ __launch_bounds__(256)
void attn_apply_kernel(const float* __restrict__ attn,
                       const float* __restrict__ qkvdw,
                       float* __restrict__ out) {
    const int bh = blockIdx.y;
    const int b = bh >> 3, head = bh & 7;
    const float* vp = qkvdw + ((size_t)b * 576 + 384 + head * 24) * HW;

    __shared__ float As[576];
    for (int i = threadIdx.x; i < 576; i += 256)
        As[i] = attn[(size_t)bh * 1024 + (i / 24) * 32 + (i % 24)];
    __syncthreads();

    const int n = blockIdx.x * 512 + threadIdx.x * 2;
    f32x2 acc[24] = {};
    for (int d = 0; d < 24; d++) {
        const f32x2 vv = *(const f32x2*)(vp + (size_t)d * HW + n);
#pragma unroll
        for (int c = 0; c < 24; c++) {
            acc[c][0] += As[c * 24 + d] * vv[0];
            acc[c][1] += As[c * 24 + d] * vv[1];
        }
    }
    float* op = out + ((size_t)b * 192 + head * 24) * HW + n;
#pragma unroll
    for (int c = 0; c < 24; c++) *(f32x2*)(op + (size_t)c * HW) = acc[c];
}

// ---------------------------------------------------------------------------
extern "C" void kernel_launch(void* const* d_in, const int* in_sizes, int n_in,
                              void* d_out, int out_size, void* d_ws,
                              size_t ws_size, hipStream_t stream) {
    const float* x      = (const float*)d_in[0];
    const float* w_qkv  = (const float*)d_in[1];
    const float* w_dw   = (const float*)d_in[2];
    const float* w_proj = (const float*)d_in[3];
    const float* temp   = (const float*)d_in[4];
    float* out = (float*)d_out;

    float* ws = (float*)d_ws;
    const size_t QKV = (size_t)8 * 576 * HW;   // 75,497,472 floats
    float* qkv      = ws;                       // [8][576][HW]
    float* qkvdw    = ws + QKV;                 // [8][576][HW]
    float* scales   = ws + 2 * QKV;             // 3072 floats (q then k)
    float* attnb    = scales + 4096;            // [64][32][32]
    float* attn_out = qkv;                      // alias: qkv dead after dwconv

    // 1) qkv = w_qkv * x          (bf16 WMMA GEMM + TDM staging, O=576)
    gemm_wmma_bf16<192><<<dim3(128, 9, 8), 256, 0, stream>>>(w_qkv, x, qkv);
    // 2) depthwise dilated conv
    dwconv_kernel<<<dim3(64, 4608), 256, 0, stream>>>(qkv, w_dw, qkvdw);
    // 3) L2-norm scales for q and k
    norm_kernel<<<dim3(3072), 256, 0, stream>>>(qkvdw, scales);
    // 4) attn = softmax(q k^T * temperature)   (bf16 WMMA, K=16384)
    attn_softmax_kernel<<<dim3(64), 256, 0, stream>>>(qkvdw, scales,
                                                      scales + 1536, temp, attnb);
    // 5) attn_out = attn * v
    attn_apply_kernel<<<dim3(32, 64), 256, 0, stream>>>(attnb, qkvdw, attn_out);
    // 6) out = w_proj * attn_out  (bf16 WMMA GEMM, O=192)
    gemm_wmma_bf16<192><<<dim3(128, 3, 8), 256, 0, stream>>>(w_proj, attn_out, out);

    (void)in_sizes; (void)n_in; (void)out_size; (void)ws_size;
}
